// SelfAttention_73280732004754
// MI455X (gfx1250) — compile-verified, matrix operand order checked by reading, not run on hardware
//
#include <hip/hip_runtime.h>
#include <stdint.h>

// ---------------- problem constants ----------------
#define DIM_   2048
#define NH_    16
#define HD_    128      // DIM_/NH_
#define LC_    64       // HD_/2
#define BB_    2
#define SS_    2048
#define MROWS_ (BB_ * SS_)   // 4096
#define EPS_   1e-6f

// GEMM tiling: WG = 8 waves -> 128(M) x 256(N) tile, wave tile 64x64.
#define TM 128
#define TN 256
#define TK 32

typedef __attribute__((ext_vector_type(16))) __bf16 v16bf;
typedef __attribute__((ext_vector_type(8)))  __bf16 v8bf;
typedef __attribute__((ext_vector_type(8)))  float  v8f;
typedef __attribute__((ext_vector_type(4)))  unsigned int v4u;
typedef __attribute__((ext_vector_type(4)))  int v4i;
typedef __attribute__((ext_vector_type(8)))  int v8i;

// ---------------- bf16 helpers (RNE, bit-exact, no libcalls) ----------------
__device__ __forceinline__ __bf16 f2bf(float f) {
  union { float f; uint32_t u; } a; a.f = f;
  uint32_t r = a.u + 0x7FFFu + ((a.u >> 16) & 1u);
  union { uint16_t s; __bf16 b; } o; o.s = (uint16_t)(r >> 16);
  return o.b;
}
__device__ __forceinline__ float bf2f(__bf16 b) {
  union { uint16_t s; __bf16 b; } i; i.b = b;
  union { float f; uint32_t u; } o; o.u = ((uint32_t)i.s) << 16;
  return o.f;
}

// ---------------- WMMA wrapper ----------------
__device__ __forceinline__ v8f wmma_bf16(v16bf a, v16bf b, v8f c) {
#if defined(__gfx1250__)
  return __builtin_amdgcn_wmma_f32_16x16x32_bf16(
      /*neg_a=*/false, a, /*neg_b=*/false, b,
      /*c_mod=*/(short)0, c, /*reuse_a=*/false, /*reuse_b=*/false);
#else
  return c; // host-pass parse only
#endif
}

// 16-bit operand fragment loader (ISA 7.12.2 A-matrix 16x32 layout; B mirrored
// with lane = N). Per lane: row = row_base + (lane&15); K chunks
// {0..7,16..23} for lanes 0-15 and {8..15,24..31} for lanes 16-31.
__device__ __forceinline__ v16bf load_frag(const __bf16* base, size_t ld,
                                           int row_base, int k_base) {
  int lane = (int)(threadIdx.x & 31u);
  const __bf16* p = base + (size_t)(row_base + (lane & 15)) * ld
                         + (size_t)(k_base + ((lane >> 4) << 3));
  v8bf lo = *(const v8bf*)p;         // K = kb .. kb+7
  v8bf hi = *(const v8bf*)(p + 16);  // K = kb+16 .. kb+23
  return __builtin_shufflevector(lo, hi, 0,1,2,3,4,5,6,7,8,9,10,11,12,13,14,15);
}

// ---------------- Tensor Data Mover: 2D tile -> LDS ----------------
#if __has_builtin(__builtin_amdgcn_tensor_load_to_lds)
#define HAVE_TDM 1
// Copy tile_d1 rows of tile_d0 bf16 elements (row stride = stride0 elements)
// from global into contiguous LDS. Issued by one lane; EXEC ignored by tensor
// ops; tracked with TENSORcnt (in order per wave -> usable for double buffer).
__device__ __forceinline__ void tdm_load_2d(void* lds_dst, const void* gsrc,
                                            unsigned tile_d0, unsigned tile_d1,
                                            unsigned long long stride0) {
  unsigned lds = (unsigned)(uintptr_t)lds_dst;      // low 32 bits = LDS offset
  unsigned long long ga = (unsigned long long)(uintptr_t)gsrc;
  v4u g0;
  g0[0] = 1u;                                        // count=1, user descriptor
  g0[1] = lds;                                       // lds_addr
  g0[2] = (unsigned)(ga & 0xFFFFFFFFull);            // global_addr[31:0]
  g0[3] = (unsigned)((ga >> 32) & 0x1FFFFFFull) | (2u << 30); // addr[56:32], type=2
  unsigned td0 = (unsigned)stride0;                  // tensor_dim0 (row length)
  unsigned td1 = 0x40000000u;                        // tensor_dim1: huge, in-bounds
  v8i g1;
  g1[0] = (int)(1u << 16);                                           // data_size=2B
  g1[1] = (int)((td0 & 0xFFFFu) << 16);                              // dim0 lo16
  g1[2] = (int)(((td0 >> 16) & 0xFFFFu) | ((td1 & 0xFFFFu) << 16));  // dim0 hi/dim1 lo
  g1[3] = (int)(((td1 >> 16) & 0xFFFFu) | ((tile_d0 & 0xFFFFu) << 16));
  g1[4] = (int)(tile_d1 & 0xFFFFu);                                  // tile_dim1
  g1[5] = (int)(unsigned)(stride0 & 0xFFFFFFFFull);                  // dim0_stride lo
  g1[6] = (int)(unsigned)((stride0 >> 32) & 0xFFFFull);              // dim0_stride hi
  g1[7] = 0;
  v4i z4 = {0, 0, 0, 0};
#if __clang_major__ >= 23
  v8i z8 = {0, 0, 0, 0, 0, 0, 0, 0};
  __builtin_amdgcn_tensor_load_to_lds(g0, g1, z4, z4, z8, 0);
#else
  __builtin_amdgcn_tensor_load_to_lds(g0, g1, z4, z4, 0);
#endif
}
#else
#define HAVE_TDM 0
#endif

__device__ __forceinline__ void wait_tdm_le2() {
#if __has_builtin(__builtin_amdgcn_s_wait_tensorcnt)
  __builtin_amdgcn_s_wait_tensorcnt(2);
#endif
}
__device__ __forceinline__ void wait_tdm_le1() {
#if __has_builtin(__builtin_amdgcn_s_wait_tensorcnt)
  __builtin_amdgcn_s_wait_tensorcnt(1);
#endif
}
__device__ __forceinline__ void wait_tdm0() {
#if __has_builtin(__builtin_amdgcn_s_wait_tensorcnt)
  __builtin_amdgcn_s_wait_tensorcnt(0);
#endif
}

// ---------------- half-wave (16-lane group) reductions ----------------
__device__ __forceinline__ float half16_max(float v) {
#pragma unroll
  for (int off = 1; off < 16; off <<= 1) v = fmaxf(v, __shfl_xor(v, off, 32));
  return v;
}
__device__ __forceinline__ float half16_sum(float v) {
#pragma unroll
  for (int off = 1; off < 16; off <<= 1) v += __shfl_xor(v, off, 32);
  return v;
}

// ---------------- kernel: fp32 -> bf16 convert ----------------
__global__ __launch_bounds__(256) void cvt_f32_to_bf16(
    const float* __restrict__ in, __bf16* __restrict__ out, int n) {
  int i = blockIdx.x * blockDim.x + threadIdx.x;
  int stride = gridDim.x * blockDim.x;
  for (; i < n; i += stride) {
    __builtin_prefetch(in + i + stride, 0, 0);  // global_prefetch_b8
    out[i] = f2bf(in[i]);
  }
}

// ---------------- kernel: C = A * W^T + bias, bf16 WMMA ----------------
// A: [M][K] bf16 row-major, W: [N][K] bf16 row-major. TDM double-buffers the
// K-slabs: slab k+1 is in flight (TENSORcnt<=2) while slab k is consumed.
// 8 waves: 2 along M x 4 along N; wave tile 64x64 -> 16 WMMA / 16 ds_load_b128.
__global__ __launch_bounds__(256) void gemm_wmma_bf16(
    const __bf16* __restrict__ A, const __bf16* __restrict__ W,
    const float* __restrict__ bias,
    float* __restrict__ Cf, __bf16* __restrict__ Cb,
    int M, int N, int K) {
  __shared__ __bf16 As[2][TM][TK];  // 2 x 8 KB
  __shared__ __bf16 Ws[2][TN][TK];  // 2 x 16 KB
  const int wave  = (int)(threadIdx.x >> 5);
  const int m_off = (wave & 1) * 64;   // 2 waves along M
  const int n_off = (wave >> 1) * 64;  // 4 waves along N
  const int lane  = (int)(threadIdx.x & 31u);
  const int mblk  = blockIdx.y * TM;
  const int nblk  = blockIdx.x * TN;

  const v8f z8f = {0, 0, 0, 0, 0, 0, 0, 0};
  v8f acc[4][4];
#pragma unroll
  for (int i = 0; i < 4; ++i)
#pragma unroll
    for (int j = 0; j < 4; ++j) acc[i][j] = z8f;

#if HAVE_TDM
  if (threadIdx.x == 0) {  // prologue: stage slab 0 into buffer 0
    tdm_load_2d(&As[0][0][0], A + (size_t)mblk * K, TK, TM, (unsigned long long)K);
    tdm_load_2d(&Ws[0][0][0], W + (size_t)nblk * K, TK, TN, (unsigned long long)K);
  }
#endif

  for (int kk = 0; kk < K; kk += TK) {
    const int cur = (kk >> 5) & 1;
    const bool has_next = (kk + TK) < K;
#if HAVE_TDM
    if (has_next && threadIdx.x == 0) {  // prefetch slab k+1 into other buffer
      tdm_load_2d(&As[cur ^ 1][0][0], A + (size_t)mblk * K + kk + TK, TK, TM,
                  (unsigned long long)K);
      tdm_load_2d(&Ws[cur ^ 1][0][0], W + (size_t)nblk * K + kk + TK, TK, TN,
                  (unsigned long long)K);
    }
    if (wave == 0) {  // in-order TENSORcnt: <=2 means current pair landed
      if (has_next) wait_tdm_le2(); else wait_tdm0();
    }
#else
    for (int i = (int)threadIdx.x; i < TN * (TK / 8); i += 256) {
      int r = i / (TK / 8), c = (i % (TK / 8)) * 8;
      if (r < TM)
        *(v8bf*)&As[cur][r][c] = *(const v8bf*)(A + (size_t)(mblk + r) * K + kk + c);
      *(v8bf*)&Ws[cur][r][c] = *(const v8bf*)(W + (size_t)(nblk + r) * K + kk + c);
    }
#endif
    __syncthreads();  // data ready

    v16bf af[4], bfr[4];
#pragma unroll
    for (int i = 0; i < 4; ++i)
      af[i] = load_frag(&As[cur][0][0], TK, m_off + i * 16, 0);
#pragma unroll
    for (int j = 0; j < 4; ++j)
      bfr[j] = load_frag(&Ws[cur][0][0], TK, n_off + j * 16, 0);
#pragma unroll
    for (int i = 0; i < 4; ++i)
#pragma unroll
      for (int j = 0; j < 4; ++j) acc[i][j] = wmma_bf16(af[i], bfr[j], acc[i][j]);

    __syncthreads();  // compute done; next iter may refill this buffer
  }

  // C/D layout: VGPR r -> rows {r, r+8}, lane halves; lane%16 = column.
#pragma unroll
  for (int i = 0; i < 4; ++i)
#pragma unroll
    for (int j = 0; j < 4; ++j) {
      int col  = nblk + n_off + j * 16 + (lane & 15);
      int row0 = mblk + m_off + i * 16 + ((lane >> 4) << 3);
      float bb = bias[col];
#pragma unroll
      for (int r = 0; r < 8; ++r) {
        float v = acc[i][j][r] + bb;
        if (Cb) Cb[(size_t)(row0 + r) * N + col] = f2bf(v);
        else    Cf[(size_t)(row0 + r) * N + col] = v;
      }
    }
}

// ---------------- kernel: fused RMSNorm + RoPE, in place on bf16 ----------------
__global__ __launch_bounds__(256) void rmsnorm_rope(
    __bf16* __restrict__ Y, const float* __restrict__ g,
    const float* __restrict__ freqs) {
  const int row = blockIdx.x;
  const int s   = row & (SS_ - 1);
  __bf16* y = Y + (size_t)row * DIM_;

  float ssum = 0.f;
  for (int d = (int)threadIdx.x; d < DIM_; d += 256) {
    float v = bf2f(y[d]);
    ssum += v * v;
  }
#pragma unroll
  for (int off = 16; off; off >>= 1) ssum += __shfl_xor(ssum, off, 32);
  __shared__ float red[8];
  if ((threadIdx.x & 31) == 0) red[threadIdx.x >> 5] = ssum;
  __syncthreads();  // also fences all reads of y before in-place writes
  float tot = red[0] + red[1] + red[2] + red[3] + red[4] + red[5] + red[6] + red[7];
  float inv = rsqrtf(tot * (1.0f / DIM_) + EPS_);

  for (int p = (int)threadIdx.x; p < DIM_ / 2; p += 256) {
    int d0 = p * 2;
    float yr = bf2f(y[d0])     * inv * g[d0];
    float yi = bf2f(y[d0 + 1]) * inv * g[d0 + 1];
    int lc = p & (LC_ - 1);  // (d % HEAD_DIM)/2
    float cr = freqs[((size_t)s * LC_ + lc) * 2 + 0];
    float ci = freqs[((size_t)s * LC_ + lc) * 2 + 1];
    y[d0]     = f2bf(yr * cr - yi * ci);
    y[d0 + 1] = f2bf(yr * ci + yi * cr);
  }
}

// ---------------- kernel: flash attention, bf16 WMMA, pipelined staging -------
// grid = (S/64, NUM_HEADS, B), block = 128 (4 waves). Wave owns 16 query rows.
// K blocks double-buffered via TDM; V blocks register-pipelined + transposed.
__global__ __launch_bounds__(128) void attn_wmma(
    const __bf16* __restrict__ Q, const __bf16* __restrict__ Kb,
    const __bf16* __restrict__ V, __bf16* __restrict__ O) {
  __shared__ __bf16 Ks[2][32][HD_];    // [kv][d]        2 x 8 KB
  __shared__ __bf16 Vt[2][HD_][32];    // [d][kv]        2 x 8 KB
  __shared__ __bf16 Ps[4][16][32];     // per-wave P     4 KB

  const int tid  = (int)threadIdx.x;
  const int wave = tid >> 5;
  const int lane = tid & 31;
  const int b = blockIdx.z, h = blockIdx.y;
  const int q0 = blockIdx.x * 64 + wave * 16;
  const size_t base = (size_t)b * SS_ * DIM_ + (size_t)h * HD_;
  const float scale = 0.08838834764831845f;  // 1/sqrt(128)

  // Q fragments: 16 rows x 128 head-dim, resident in VGPRs for the whole pass.
  v16bf qf[4];
#pragma unroll
  for (int c = 0; c < 4; ++c)
    qf[c] = load_frag(Q + base + (size_t)q0 * DIM_, DIM_, 0, c * 32);

  const v8f z8f = {0, 0, 0, 0, 0, 0, 0, 0};
  v8f o[8];
#pragma unroll
  for (int t = 0; t < 8; ++t) o[t] = z8f;
  float mi[8], li[8];
#pragma unroll
  for (int r = 0; r < 8; ++r) { mi[r] = -1.0e30f; li[r] = 0.f; }

  // V staging map: thread -> row vr = tid/4 (0..31), cols vc0..vc0+31.
  const int vr  = tid >> 2;
  const int vc0 = (tid & 3) * 32;
  v8bf vreg[4];

  // ---- prologue: stage kv-block 0 ----
#if HAVE_TDM
  if (tid == 0) tdm_load_2d(&Ks[0][0][0], Kb + base, HD_, 32,
                            (unsigned long long)DIM_);
#else
  for (int i = tid; i < 32 * (HD_ / 8); i += 128) {
    int r = i / (HD_ / 8), c = (i % (HD_ / 8)) * 8;
    *(v8bf*)&Ks[0][r][c] = *(const v8bf*)(Kb + base + (size_t)r * DIM_ + c);
  }
#endif
#pragma unroll
  for (int j = 0; j < 4; ++j)
    vreg[j] = *(const v8bf*)(V + base + (size_t)vr * DIM_ + vc0 + j * 8);

  const int nblk = SS_ / 32;
  for (int i = 0; i < nblk; ++i) {
    const int cur = i & 1;
    const bool has_next = (i + 1) < nblk;

    // commit V block i (loaded during previous iteration) transposed into LDS
#pragma unroll
    for (int j = 0; j < 4; ++j)
#pragma unroll
      for (int e = 0; e < 8; ++e) Vt[cur][vc0 + j * 8 + e][vr] = vreg[j][e];

    // prefetch block i+1: K via TDM, V into registers (latency hidden by compute)
    if (has_next) {
#if HAVE_TDM
      if (tid == 0)
        tdm_load_2d(&Ks[cur ^ 1][0][0],
                    Kb + base + (size_t)(i + 1) * 32 * DIM_, HD_, 32,
                    (unsigned long long)DIM_);
#else
      for (int t = tid; t < 32 * (HD_ / 8); t += 128) {
        int r = t / (HD_ / 8), c = (t % (HD_ / 8)) * 8;
        *(v8bf*)&Ks[cur ^ 1][r][c] =
            *(const v8bf*)(Kb + base + (size_t)((i + 1) * 32 + r) * DIM_ + c);
      }
#endif
#pragma unroll
      for (int j = 0; j < 4; ++j)
        vreg[j] = *(const v8bf*)(V + base +
                                 (size_t)((i + 1) * 32 + vr) * DIM_ + vc0 + j * 8);
    }
#if HAVE_TDM
    if (wave == 0) { if (has_next) wait_tdm_le1(); else wait_tdm0(); }
#endif
    __syncthreads();  // Ks[cur]/Vt[cur] ready for all waves

    // S = Q * K^T : two 16x16 n-subtiles over this 32-wide kv block.
    v8f s0 = z8f, s1 = z8f;
#pragma unroll
    for (int c = 0; c < 4; ++c) {
      s0 = wmma_bf16(qf[c], load_frag(&Ks[cur][0][0], HD_,  0, c * 32), s0);
      s1 = wmma_bf16(qf[c], load_frag(&Ks[cur][0][0], HD_, 16, c * 32), s1);
    }

    // Online softmax per row (row r = element r across 16-lane halves).
    float p0[8], p1[8];
#pragma unroll
    for (int r = 0; r < 8; ++r) {
      float a = s0[r] * scale, bs = s1[r] * scale;
      float mx = half16_max(fmaxf(a, bs));
      float mnew = fmaxf(mi[r], mx);
      float corr = __expf(mi[r] - mnew);
      float e0 = __expf(a - mnew), e1 = __expf(bs - mnew);
      float rs = half16_sum(e0 + e1);
      li[r] = li[r] * corr + rs;
      mi[r] = mnew;
      p0[r] = e0; p1[r] = e1;
#pragma unroll
      for (int t = 0; t < 8; ++t) o[t][r] *= corr;  // rescale O rows
    }

    // C-layout -> A-fragment layout via per-wave LDS round-trip.
    int prow = (lane >> 4) << 3;
    int pcol = lane & 15;
#pragma unroll
    for (int r = 0; r < 8; ++r) {
      Ps[wave][prow + r][pcol]      = f2bf(p0[r]);
      Ps[wave][prow + r][pcol + 16] = f2bf(p1[r]);
    }
    v16bf pa = load_frag(&Ps[wave][0][0], 32, 0, 0);
#pragma unroll
    for (int t = 0; t < 8; ++t)
      o[t] = wmma_bf16(pa, load_frag(&Vt[cur][0][0], 32, t * 16, 0), o[t]);

    __syncthreads();  // all waves done with Ks[cur]/Vt[cur]
  }

  // epilogue: O /= l, store bf16 into attn buffer [B][S][DIM]
  int col0 = lane & 15;
  int row0 = (lane >> 4) << 3;
#pragma unroll
  for (int r = 0; r < 8; ++r) {
    float invl = 1.0f / li[r];
#pragma unroll
    for (int t = 0; t < 8; ++t)
      O[base + (size_t)(q0 + row0 + r) * DIM_ + t * 16 + col0] =
          f2bf(o[t][r] * invl);
  }
}

// ---------------- host launcher ----------------
extern "C" void kernel_launch(void* const* d_in, const int* in_sizes, int n_in,
                              void* d_out, int out_size, void* d_ws, size_t ws_size,
                              hipStream_t stream) {
  const float* x  = (const float*)d_in[0];
  const float* fr = (const float*)d_in[1];
  const float* wq = (const float*)d_in[2];
  const float* bq = (const float*)d_in[3];
  const float* wk = (const float*)d_in[4];
  const float* bk = (const float*)d_in[5];
  const float* wv = (const float*)d_in[6];
  const float* bv = (const float*)d_in[7];
  const float* wo = (const float*)d_in[8];
  const float* bo = (const float*)d_in[9];
  const float* gq = (const float*)d_in[10];
  const float* gk = (const float*)d_in[11];
  float* out = (float*)d_out;

  char* ws = (char*)d_ws;
  size_t off = 0;
  auto alloc = [&](size_t bytes) -> char* {
    char* p = ws + off;
    off += (bytes + 255) & ~(size_t)255;
    return p;
  };
  const size_t XB = (size_t)MROWS_ * DIM_ * 2;  // bf16 activations
  const size_t WB = (size_t)DIM_ * DIM_ * 2;    // bf16 weights
  __bf16* xb  = (__bf16*)alloc(XB);
  __bf16* wqb = (__bf16*)alloc(WB);
  __bf16* wkb = (__bf16*)alloc(WB);
  __bf16* wvb = (__bf16*)alloc(WB);
  __bf16* wob = (__bf16*)alloc(WB);
  __bf16* qb  = (__bf16*)alloc(XB);
  __bf16* kb  = (__bf16*)alloc(XB);
  __bf16* vb  = (__bf16*)alloc(XB);
  __bf16* ab  = (__bf16*)alloc(XB);

  // 1) convert activations + weights to bf16
  cvt_f32_to_bf16<<<1024, 256, 0, stream>>>(x,  xb,  MROWS_ * DIM_);
  cvt_f32_to_bf16<<<1024, 256, 0, stream>>>(wq, wqb, DIM_ * DIM_);
  cvt_f32_to_bf16<<<1024, 256, 0, stream>>>(wk, wkb, DIM_ * DIM_);
  cvt_f32_to_bf16<<<1024, 256, 0, stream>>>(wv, wvb, DIM_ * DIM_);
  cvt_f32_to_bf16<<<1024, 256, 0, stream>>>(wo, wob, DIM_ * DIM_);

  // 2) QKV projections (bf16 out, bias fused)
  dim3 ggrid(DIM_ / TN, MROWS_ / TM);  // (8, 32)
  gemm_wmma_bf16<<<ggrid, 256, 0, stream>>>(xb, wqb, bq, nullptr, qb,
                                            MROWS_, DIM_, DIM_);
  gemm_wmma_bf16<<<ggrid, 256, 0, stream>>>(xb, wkb, bk, nullptr, kb,
                                            MROWS_, DIM_, DIM_);
  gemm_wmma_bf16<<<ggrid, 256, 0, stream>>>(xb, wvb, bv, nullptr, vb,
                                            MROWS_, DIM_, DIM_);

  // 3) RMSNorm + RoPE on Q and K (in place)
  rmsnorm_rope<<<MROWS_, 256, 0, stream>>>(qb, gq, fr);
  rmsnorm_rope<<<MROWS_, 256, 0, stream>>>(kb, gk, fr);

  // 4) flash attention
  attn_wmma<<<dim3(SS_ / 64, NH_, BB_), 128, 0, stream>>>(qb, kb, vb, ab);

  // 5) output projection, fp32 straight into d_out
  gemm_wmma_bf16<<<ggrid, 256, 0, stream>>>(ab, wob, bo, out, nullptr,
                                            MROWS_, DIM_, DIM_);
}
